// EquivariantEdgeAttention_25924422599458
// MI455X (gfx1250) — matrix operand order repaired and checked by reading, not compiled, not run on hardware
//
#include <hip/hip_runtime.h>

#define E_N 384
#define HID 128
#define NH  8
#define DH  16
#define NR  64

typedef __attribute__((ext_vector_type(16))) _Float16 v16h;
typedef __attribute__((ext_vector_type(8)))  _Float16 v8h;
typedef __attribute__((ext_vector_type(8)))  float    v8f;

__device__ __forceinline__ v8f wmma_f16(v16h a, v16h b, v8f c) {
  return __builtin_amdgcn_wmma_f32_16x16x32_f16(false, a, false, b, (short)0, c, false, false);
}

__device__ __forceinline__ float fast_rcp(float x) {
  return __builtin_amdgcn_rcpf(x);
}

__device__ __forceinline__ float siluf(float x) {
  // x * sigmoid(x) using hardware rcp (avoids IEEE divide expansion)
  return x * fast_rcp(1.0f + __expf(-x));
}

__device__ __forceinline__ float sigmoidf_(float x) {
  return fast_rcp(1.0f + __expf(-x));
}

// Build A fragment from row-major f16 LDS tile [rows][128]
__device__ __forceinline__ v16h lds_afrag(const _Float16* sX, int row, int kk, int ko) {
  const v8h lo = *(const v8h*)(sX + row * HID + kk + ko);
  const v8h hi = *(const v8h*)(sX + row * HID + kk + ko + 8);
  return __builtin_shufflevector(lo, hi, 0,1,2,3,4,5,6,7,8,9,10,11,12,13,14,15);
}

// ---------------- K0: Q/K/V projections (384x128 @ 128x128 x3) ----------------
__global__ __launch_bounds__(128) void qkv_kernel(
    const float* __restrict__ ef, const float* __restrict__ wq,
    const float* __restrict__ wk, const float* __restrict__ wv,
    float* __restrict__ q, float* __restrict__ k, float* __restrict__ v) {
  __shared__ float sE[HID];
  const int i = blockIdx.x, c = threadIdx.x;
  sE[c] = ef[i * HID + c];
  __syncthreads();
  float aq = 0.f, ak = 0.f, av = 0.f;
  for (int d = 0; d < HID; ++d) {
    const float e = sE[d];
    aq += e * wq[d * HID + c];
    ak += e * wk[d * HID + c];
    av += e * wv[d * HID + c];
  }
  q[i * HID + c] = aq;
  k[i * HID + c] = ak;
  v[i * HID + c] = av;
}

// ---------------- K1: weight transpose + f16 convert ----------------
__global__ __launch_bounds__(128) void wconv_kernel(
    const float* __restrict__ a_w1, const float* __restrict__ a_w2,
    const float* __restrict__ a_w3,
    _Float16* __restrict__ w2t, _Float16* __restrict__ wrbft,
    _Float16* __restrict__ w3t) {
  const int n = blockIdx.x, kk = threadIdx.x;
  w2t[n * HID + kk] = (_Float16)a_w2[kk * HID + n];
  if (kk < NR) wrbft[n * NR + kk] = (_Float16)a_w1[(32 + kk) * HID + n];
  if (n < 16)  w3t[n * HID + kk] = (n < NH) ? (_Float16)a_w3[kk * NH + n] : (_Float16)0.0f;
}

// ---------------- K2: per-(head,node) contributions qc/kc + gates ----------------
__global__ __launch_bounds__(128) void qckc_kernel(
    const float* __restrict__ q, const float* __restrict__ k,
    const float* __restrict__ v, const float* __restrict__ a_w1,
    const float* __restrict__ g_w1, const float* __restrict__ g_b1,
    const float* __restrict__ g_w2, const float* __restrict__ g_b2,
    float* __restrict__ qc, float* __restrict__ kc, float* __restrict__ gate) {
  __shared__ float red[HID];
  const int i = blockIdx.x, h = blockIdx.y, c = threadIdx.x;
  float qv = 0.f, kv = 0.f, gv = 0.f;
  for (int d = 0; d < DH; ++d) {
    const float qd = q[i * HID + h * DH + d];
    const float kd = k[i * HID + h * DH + d];
    const float vd = v[i * HID + h * DH + d];
    qv += qd * a_w1[d * HID + c];
    kv += kd * a_w1[(DH + d) * HID + c];
    gv += vd * g_w1[d * HID + c];
  }
  qc[(h * E_N + i) * HID + c] = qv;
  kc[(h * E_N + i) * HID + c] = kv;
  red[c] = siluf(gv + g_b1[c]) * g_w2[c];
  __syncthreads();
  for (int s = 64; s > 0; s >>= 1) {
    if (c < s) red[c] += red[c + s];
    __syncthreads();
  }
  if (c == 0) gate[h * E_N + i] = sigmoidf_(red[0] + g_b2[0]);
}

// ---------------- K3: fused pairwise MLP (WMMA) -> scores ----------------
// grid: (24 i-tiles, 8 heads, 6 j-blocks of 4 tiles). block = 128 (4 waves).
__global__ __launch_bounds__(128) void pair_mlp_kernel(
    const float* __restrict__ coords, const float* __restrict__ emask,
    const float* __restrict__ rbf_c, const float* __restrict__ rbf_w,
    const float* __restrict__ a_w1, const float* __restrict__ a_b1,
    const float* __restrict__ a_b2, const float* __restrict__ a_b3,
    const float* __restrict__ qc, const float* __restrict__ kc,
    const _Float16* __restrict__ wrbft, const _Float16* __restrict__ w2t,
    const _Float16* __restrict__ w3t, float* __restrict__ S) {
  // Exactly 64KB static LDS total
  __shared__ __attribute__((aligned(16))) _Float16 sWR[HID * NR];   // 16KB  B weights layer1 [N][64]
  __shared__ __attribute__((aligned(16))) _Float16 sW2[HID * HID];  // 32KB  B weights layer2 [N][128]
  __shared__ __attribute__((aligned(16))) _Float16 sX1[64 * HID];   // 16KB  x1/x2 staging, 16 rows/wave

  const int i0  = blockIdx.x * 16;
  const int h   = blockIdx.y;
  const int lane = threadIdx.x & 31;
  const int wv   = threadIdx.x >> 5;
  const int nB   = lane & 15;          // N index (B/C frags) & A-row M index
  const int hi   = lane >> 4;          // lane half
  const int ko   = hi * 16;            // A-frag K sub-offset
  const int kB   = hi * 16;            // B-frag K sub-offset
  const int rbase = wv * 16;           // wave-private sX1 row base

  // ---- stage layer-1/2 B weights into LDS (once per block) ----
  {
    const uint4* srcR = (const uint4*)wrbft;  // 16KB = 1024 uint4
    uint4* dstR = (uint4*)sWR;
    for (int idx = threadIdx.x; idx < 1024; idx += 128) dstR[idx] = srcR[idx];
    const uint4* src2 = (const uint4*)w2t;    // 32KB = 2048 uint4
    uint4* dst2 = (uint4*)sW2;
    for (int idx = threadIdx.x; idx < 2048; idx += 128) dst2[idx] = src2[idx];
  }
  __syncthreads();

  const float b3h = a_b3[h];

  for (int jt = 0; jt < 4; ++jt) {
    const int j0 = (blockIdx.z * 4 + jt) * 16;

    for (int t = 0; t < 4; ++t) {
      const int m    = wv * 4 + t;     // i-row within tile
      const int iRow = i0 + m;

      // ---- pair geometry for this lane's A-row pair (jj = nB) ----
      const float ci0 = coords[iRow * 3 + 0];
      const float ci1 = coords[iRow * 3 + 1];
      const float ci2 = coords[iRow * 3 + 2];
      const int jA = j0 + nB;
      const float dx = ci0 - coords[jA * 3 + 0];
      const float dy = ci1 - coords[jA * 3 + 1];
      const float dz = ci2 - coords[jA * 3 + 2];
      float dist = sqrtf(dx * dx + dy * dy + dz * dz) + 1e-8f;
      dist = fminf(fmaxf(dist, 1e-8f), 1e8f);
      const float rcut = (dist <= 10.0f) ? 1.0f : 0.0f;

      auto rbfv = [&](int kr) -> _Float16 {
        const float dc = dist - rbf_c[kr];
        return (_Float16)(__expf(-rbf_w[kr] * dc * dc) * rcut);
      };

      // ---- layer-1 A fragments (rbf), built in registers in contiguous k = 16*hi + s order ----
      v16h a0, a1;
      #pragma unroll
      for (int t8 = 0; t8 < 8; ++t8) {
        a0[t8]     = rbfv(ko + t8);
        a0[8 + t8] = rbfv(8 + ko + t8);
        a1[t8]     = rbfv(32 + ko + t8);
        a1[8 + t8] = rbfv(40 + ko + t8);
      }

      // dot(i, j) for the epilogue's M positions (M = g + hi*8)
      const float* cjb = coords + (j0 + hi * 8) * 3;
      float dotE[8];
      #pragma unroll
      for (int g = 0; g < 8; ++g) {
        float dd = ci0 * cjb[g * 3 + 0] + ci1 * cjb[g * 3 + 1] +
                   ci2 * cjb[g * 3 + 2];
        dotE[g] = fminf(fmaxf(dd, -1e8f), 1e8f);
      }

      const float* qcb  = qc + (h * E_N + iRow) * HID;
      const float* kcb0 = kc + (h * E_N + j0 + hi * 8) * HID;

      // ---- layer 1: x1 = silu(rbf@Wr + qc + kc + dot*w96 + b1) ----
      #pragma unroll 2
      for (int nn = 0; nn < 8; ++nn) {
        v8f acc = {};
        const _Float16* wb = sWR + (nn * 16 + nB) * NR + kB;
        acc = wmma_f16(a0, *(const v16h*)(wb), acc);
        acc = wmma_f16(a1, *(const v16h*)(wb + 32), acc);
        const int c = nn * 16 + nB;
        // batch the 8 kc loads (base + immediate offsets)
        const float* kcb = kcb0 + c;
        float kcv[8];
        #pragma unroll
        for (int g = 0; g < 8; ++g) kcv[g] = kcb[g * HID];
        const float w96 = a_w1[96 * HID + c];
        const float qb  = qcb[c] + a_b1[c];
        #pragma unroll
        for (int g = 0; g < 8; ++g) {
          const float x = acc[g] + qb + kcv[g] + dotE[g] * w96;
          sX1[(rbase + g + hi * 8) * HID + c] = (_Float16)siluf(x);
        }
      }

      // ---- layer 2: x2 = silu(x1 @ W2 + b2) ----
      const int arow = rbase + nB;
      {
        const v16h A0 = lds_afrag(sX1, arow, 0, ko);
        const v16h A1 = lds_afrag(sX1, arow, 32, ko);
        const v16h A2 = lds_afrag(sX1, arow, 64, ko);
        const v16h A3 = lds_afrag(sX1, arow, 96, ko);
        #pragma unroll 2
        for (int nn = 0; nn < 8; ++nn) {
          v8f acc = {};
          const _Float16* wb = sW2 + (nn * 16 + nB) * HID + kB;
          acc = wmma_f16(A0, *(const v16h*)(wb), acc);
          acc = wmma_f16(A1, *(const v16h*)(wb + 32), acc);
          acc = wmma_f16(A2, *(const v16h*)(wb + 64), acc);
          acc = wmma_f16(A3, *(const v16h*)(wb + 96), acc);
          const int c = nn * 16 + nB;
          const float b2v = a_b2[c];
          #pragma unroll
          for (int g = 0; g < 8; ++g)
            sX1[(rbase + g + hi * 8) * HID + c] = (_Float16)siluf(acc[g] + b2v);
        }
      }

      // ---- layer 3: scores = (x2 @ W3 + b3)[:, h] ----
      v8f acc3 = {};
      {
        const v16h X0 = lds_afrag(sX1, arow, 0, ko);
        const v16h X1 = lds_afrag(sX1, arow, 32, ko);
        const v16h X2 = lds_afrag(sX1, arow, 64, ko);
        const v16h X3 = lds_afrag(sX1, arow, 96, ko);
        const _Float16* wb = w3t + nB * HID + kB;
        acc3 = wmma_f16(X0, *(const v16h*)(wb), acc3);
        acc3 = wmma_f16(X1, *(const v16h*)(wb + 32), acc3);
        acc3 = wmma_f16(X2, *(const v16h*)(wb + 64), acc3);
        acc3 = wmma_f16(X3, *(const v16h*)(wb + 96), acc3);
      }
      if (nB == h) {
        const float* emb = emask + iRow * E_N + j0 + hi * 8;
        float* Sb = S + (h * E_N + iRow) * E_N + j0 + hi * 8;
        #pragma unroll
        for (int g = 0; g < 8; ++g) {
          float sc = acc3[g] + b3h + emb[g];
          sc = fminf(fmaxf(sc, -1e9f), 1e9f);
          Sb[g] = sc;
        }
      }
    }
  }
}

// ---------------- K4: row softmax over j (per head,row) ----------------
__global__ __launch_bounds__(128) void softmax_kernel(float* __restrict__ S) {
  __shared__ float red[HID];
  const int i = blockIdx.x, h = blockIdx.y, c = threadIdx.x;
  float* row = S + (h * E_N + i) * E_N;
  float v0 = row[c], v1 = row[c + 128], v2 = row[c + 256];
  float mx = fmaxf(v0, fmaxf(v1, v2));
  red[c] = mx;
  __syncthreads();
  for (int s = 64; s > 0; s >>= 1) {
    if (c < s) red[c] = fmaxf(red[c], red[c + s]);
    __syncthreads();
  }
  mx = red[0];
  __syncthreads();
  float e0 = __expf(v0 - mx), e1 = __expf(v1 - mx), e2 = __expf(v2 - mx);
  red[c] = e0 + e1 + e2;
  __syncthreads();
  for (int s = 64; s > 0; s >>= 1) {
    if (c < s) red[c] += red[c + s];
    __syncthreads();
  }
  const float sum = red[0];
  const float inv = (sum > 0.f && sum == sum) ? fast_rcp(sum) : 0.0f;
  float a0 = e0 * inv, a1 = e1 * inv, a2 = e2 * inv;
  if (!(a0 == a0)) a0 = 0.f;
  if (!(a1 == a1)) a1 = 0.f;
  if (!(a2 == a2)) a2 = 0.f;
  row[c] = a0; row[c + 128] = a1; row[c + 256] = a2;
}

// ---------------- K5: attn@V, output proj, LayerNorm, coord update ----------------
__global__ __launch_bounds__(128) void out_kernel(
    const float* __restrict__ S, const float* __restrict__ v,
    const float* __restrict__ ef, const float* __restrict__ wo,
    const float* __restrict__ bo, const float* __restrict__ ln_g,
    const float* __restrict__ ln_b, const float* __restrict__ coords,
    const float* __restrict__ gate, float* __restrict__ out_f,
    float* __restrict__ out_c) {
  __shared__ float sU[HID];
  __shared__ float red[HID];
  const int i = blockIdx.x, c = threadIdx.x;
  const int h = c >> 4;
  const float* attn = S + (h * E_N + i) * E_N;
  float u = 0.f;
  for (int j = 0; j < E_N; ++j) u += attn[j] * v[j * HID + c];
  sU[c] = u;
  __syncthreads();
  float y = ef[i * HID + c] + bo[c];
  for (int d = 0; d < HID; ++d) y += sU[d] * wo[d * HID + c];
  red[c] = y;
  __syncthreads();
  for (int s = 64; s > 0; s >>= 1) { if (c < s) red[c] += red[c + s]; __syncthreads(); }
  const float mu = red[0] * (1.0f / HID);
  __syncthreads();
  const float dev = y - mu;
  red[c] = dev * dev;
  __syncthreads();
  for (int s = 64; s > 0; s >>= 1) { if (c < s) red[c] += red[c + s]; __syncthreads(); }
  const float var = red[0] * (1.0f / HID);
  out_f[i * HID + c] = dev * rsqrtf(var + 1e-5f) * ln_g[c] + ln_b[c];
  __syncthreads();
  const float ci0 = coords[i * 3 + 0], ci1 = coords[i * 3 + 1], ci2 = coords[i * 3 + 2];
  float p0 = 0.f, p1 = 0.f, p2 = 0.f;
  for (int j = c; j < E_N; j += 128) {
    float w = 0.f;
    #pragma unroll
    for (int hh = 0; hh < NH; ++hh)
      w += gate[hh * E_N + i] * S[(hh * E_N + i) * E_N + j];
    p0 += w * (ci0 - coords[j * 3 + 0]);
    p1 += w * (ci1 - coords[j * 3 + 1]);
    p2 += w * (ci2 - coords[j * 3 + 2]);
  }
  red[c] = p0;
  __syncthreads();
  for (int s = 64; s > 0; s >>= 1) { if (c < s) red[c] += red[c + s]; __syncthreads(); }
  if (c == 0) out_c[i * 3 + 0] = ci0 + red[0] * (1.0f / NH);
  __syncthreads();
  red[c] = p1;
  __syncthreads();
  for (int s = 64; s > 0; s >>= 1) { if (c < s) red[c] += red[c + s]; __syncthreads(); }
  if (c == 0) out_c[i * 3 + 1] = ci1 + red[0] * (1.0f / NH);
  __syncthreads();
  red[c] = p2;
  __syncthreads();
  for (int s = 64; s > 0; s >>= 1) { if (c < s) red[c] += red[c + s]; __syncthreads(); }
  if (c == 0) out_c[i * 3 + 2] = ci2 + red[0] * (1.0f / NH);
}

extern "C" void kernel_launch(void* const* d_in, const int* in_sizes, int n_in,
                              void* d_out, int out_size, void* d_ws, size_t ws_size,
                              hipStream_t stream) {
  const float* ef   = (const float*)d_in[0];
  const float* ec   = (const float*)d_in[1];
  const float* em   = (const float*)d_in[2];
  const float* wq   = (const float*)d_in[3];
  const float* wk   = (const float*)d_in[4];
  const float* wv   = (const float*)d_in[5];
  const float* rc   = (const float*)d_in[6];
  const float* rw   = (const float*)d_in[7];
  const float* a_w1 = (const float*)d_in[8];
  const float* a_b1 = (const float*)d_in[9];
  const float* a_w2 = (const float*)d_in[10];
  const float* a_b2 = (const float*)d_in[11];
  const float* a_w3 = (const float*)d_in[12];
  const float* a_b3 = (const float*)d_in[13];
  const float* g_w1 = (const float*)d_in[14];
  const float* g_b1 = (const float*)d_in[15];
  const float* g_w2 = (const float*)d_in[16];
  const float* g_b2 = (const float*)d_in[17];
  const float* wo   = (const float*)d_in[18];
  const float* bo   = (const float*)d_in[19];
  const float* ln_g = (const float*)d_in[20];
  const float* ln_b = (const float*)d_in[21];
  // d_in[22] = edge_index, unused by the reference math

  char* ws = (char*)d_ws;
  float* q    = (float*)(ws + 0);
  float* k    = (float*)(ws + 196608);
  float* v    = (float*)(ws + 393216);
  float* qc   = (float*)(ws + 589824);
  float* kc   = (float*)(ws + 2162688);
  float* gate = (float*)(ws + 3735552);
  float* S    = (float*)(ws + 3747840);
  _Float16* w2t   = (_Float16*)(ws + 8466432);
  _Float16* wrbft = (_Float16*)(ws + 8499200);
  _Float16* w3t   = (_Float16*)(ws + 8515584);

  float* out_f = (float*)d_out;
  float* out_c = out_f + E_N * HID;

  qkv_kernel<<<dim3(E_N), dim3(128), 0, stream>>>(ef, wq, wk, wv, q, k, v);
  wconv_kernel<<<dim3(HID), dim3(128), 0, stream>>>(a_w1, a_w2, a_w3, w2t, wrbft, w3t);
  qckc_kernel<<<dim3(E_N, NH), dim3(128), 0, stream>>>(q, k, v, a_w1, g_w1, g_b1,
                                                       g_w2, g_b2, qc, kc, gate);
  pair_mlp_kernel<<<dim3(24, NH, 6), dim3(128), 0, stream>>>(
      ec, em, rc, rw, a_w1, a_b1, a_b2, a_b3, qc, kc, wrbft, w2t, w3t, S);
  softmax_kernel<<<dim3(E_N, NH), dim3(128), 0, stream>>>(S);
  out_kernel<<<dim3(E_N), dim3(128), 0, stream>>>(S, v, ef, wo, bo, ln_g, ln_b,
                                                  ec, gate, out_f, out_c);
}
